// TIC_57294863729482
// MI455X (gfx1250) — matrix operand order repaired and checked by reading, not compiled, not run on hardware
//
#include <hip/hip_runtime.h>
#include <hip/hip_bf16.h>
#include <math.h>

#define BATCH 512
#define TT    16
#define SSN   64
#define KC    32
#define DD    1024   // TT*SSN
#define NB    32     // LU panel width

typedef float  v2f   __attribute__((ext_vector_type(2)));
typedef float  v8f   __attribute__((ext_vector_type(8)));
typedef __bf16 v8bf  __attribute__((ext_vector_type(8)));
typedef __bf16 v16bf __attribute__((ext_vector_type(16)));

// ---------------------------------------------------------------------------
// Kernel 1: build block-Toeplitz mat[k] (f32, row-major DxD) from A[T,K,S,S],
// plus split-bf16 copies matH/matL (hi + residual) for the WMMA GEMM.
// block(i,j) = A[i-j].T if i>=j else A[j-i]
// ---------------------------------------------------------------------------
__global__ __launch_bounds__(256) void build_mat_kernel(const float* __restrict__ A,
                                                        float* __restrict__ mat,
                                                        __bf16* __restrict__ matH,
                                                        __bf16* __restrict__ matL) {
    int bid  = blockIdx.x;
    int k    = bid / (TT * TT);
    int rem  = bid % (TT * TT);
    int i    = rem / TT;
    int j    = rem % TT;
    int diff = i - j;
    int ad   = (diff >= 0) ? diff : -diff;
    const float* Ablk = A + ((size_t)ad * KC + k) * SSN * SSN;
    size_t base = (size_t)k * DD * DD + (size_t)(i * SSN) * DD + j * SSN;
    for (int idx = threadIdx.x; idx < SSN * SSN; idx += 256) {
        int r = idx >> 6, c = idx & 63;
        float v = (diff >= 0) ? Ablk[c * SSN + r] : Ablk[r * SSN + c];
        size_t o = base + (size_t)r * DD + c;
        mat[o] = v;
        __bf16 h = (__bf16)v;
        matH[o] = h;
        matL[o] = (__bf16)(v - (float)h);
    }
}

// ---------------------------------------------------------------------------
// Kernel 2: zero P1[B,K] and P2[K]
// ---------------------------------------------------------------------------
__global__ void init_kernel(float* __restrict__ P1, float* __restrict__ P2) {
    int i = blockIdx.x * blockDim.x + threadIdx.x;
    if (i < BATCH * KC) P1[i] = 0.0f;
    if (i < KC)         P2[i] = 0.0f;
}

// ---------------------------------------------------------------------------
// Kernel 3: quadratic form  P1[b,k] = dX[b,k,:] * mat[k] * dX[b,k,:]
// Split-bf16 (hi+lo): 3x v_wmma_f32_16x16x32_bf16 per 16x16x32 tile,
// ~2^-16 effective mantissa. Inner loop is pure loads + WMMA (no cvt).
// B fragments read from matH/matL rows via symmetry (mat[d][e]==mat[e][d]).
// Epilogue fuses Y . dX with half-wave shuffles + f32 atomics.
// ---------------------------------------------------------------------------
__global__ __launch_bounds__(256) void quadform_kernel(const float* __restrict__ X,
                                                       const float* __restrict__ mu,
                                                       const __bf16* __restrict__ matH,
                                                       const __bf16* __restrict__ matL,
                                                       float* __restrict__ P1) {
    __shared__ __bf16 sAh[16][512];
    __shared__ __bf16 sAl[16][512];

    int bid = blockIdx.x;
    int k   = bid >> 8;          // 256 blocks per cluster
    int rem = bid & 255;
    int b0  = (rem >> 3) * 16;   // 32 b-tiles
    int e0  = (rem & 7) * 128;   // 8 e-macrotiles
    int tid  = threadIdx.x;
    int wave = tid >> 5;
    int lane = tid & 31;
    int hi16 = lane >> 4;
    int l15  = lane & 15;
    int ew   = e0 + wave * 16;

    // B-fragment source row (symmetry): col N=l15 -> row (ew+l15); K-group base
    // per 16-bit B layout: lanes>=16 hold K=8..15,24..31 -> +8 elements.
    const __bf16* browH = matH + (size_t)k * DD * DD + (size_t)(ew + l15) * DD + hi16 * 8;
    const __bf16* browL = matL + (size_t)k * DD * DD + (size_t)(ew + l15) * DD + hi16 * 8;
    int kb = hi16 * 8;           // A-fragment K-group base per 16-bit A layout

    v8f c = {};

    for (int half = 0; half < 2; ++half) {
        int dbase = half * 512;
        // stage dX strip (rows b0..b0+15, d in [dbase,dbase+512)) as bf16 hi/lo,
        // vectorized: 8 elems per thread-chunk, ds_store_b128.
        for (int ch = tid; ch < 16 * 512 / 8; ch += 256) {
            int row = ch >> 6;
            int dl  = (ch & 63) * 8;
            int d   = dbase + dl;
            const float4* xp = (const float4*)&X[(size_t)(b0 + row) * DD + d];
            const float4* mp = (const float4*)&mu[k * SSN + (d & 63)];
            float4 x0 = xp[0], x1 = xp[1];
            float4 m0 = mp[0], m1 = mp[1];
            float xv[8] = {x0.x - m0.x, x0.y - m0.y, x0.z - m0.z, x0.w - m0.w,
                           x1.x - m1.x, x1.y - m1.y, x1.z - m1.z, x1.w - m1.w};
            v8bf h, l;
#pragma unroll
            for (int t = 0; t < 8; ++t) {
                float x = xv[t];
                __bf16 hh = (__bf16)x;
                h[t] = hh;
                l[t] = (__bf16)(x - (float)hh);
            }
            *(v8bf*)&sAh[row][dl] = h;
            *(v8bf*)&sAl[row][dl] = l;
        }
        __syncthreads();

        for (int d0 = 0; d0 < 512; d0 += 32) {
            // A fragments (16x32 bf16, ISA lane layout) from LDS
            v8bf ah0 = *(const v8bf*)&sAh[l15][d0 + kb];
            v8bf ah1 = *(const v8bf*)&sAh[l15][d0 + kb + 16];
            v8bf al0 = *(const v8bf*)&sAl[l15][d0 + kb];
            v8bf al1 = *(const v8bf*)&sAl[l15][d0 + kb + 16];
            v16bf ah = __builtin_shufflevector(ah0, ah1, 0,1,2,3,4,5,6,7,8,9,10,11,12,13,14,15);
            v16bf al = __builtin_shufflevector(al0, al1, 0,1,2,3,4,5,6,7,8,9,10,11,12,13,14,15);

            // B fragments: 2x8 bf16 (k = 8*hi16+0..7, 16+8*hi16+0..7) from matH/matL rows
            v8bf bh0 = *(const v8bf*)(browH + dbase + d0);
            v8bf bh1 = *(const v8bf*)(browH + dbase + d0 + 16);
            v8bf bl0 = *(const v8bf*)(browL + dbase + d0);
            v8bf bl1 = *(const v8bf*)(browL + dbase + d0 + 16);
            v16bf bh = __builtin_shufflevector(bh0, bh1, 0,1,2,3,4,5,6,7,8,9,10,11,12,13,14,15);
            v16bf bl = __builtin_shufflevector(bl0, bl1, 0,1,2,3,4,5,6,7,8,9,10,11,12,13,14,15);

            // split product: hi*hi + hi*lo + lo*hi (f32 accumulate)
            c = __builtin_amdgcn_wmma_f32_16x16x32_bf16(false, ah, false, bh, (short)0, c, false, false);
            c = __builtin_amdgcn_wmma_f32_16x16x32_bf16(false, ah, false, bl, (short)0, c, false, false);
            c = __builtin_amdgcn_wmma_f32_16x16x32_bf16(false, al, false, bh, (short)0, c, false, false);
        }
        __syncthreads();
    }

    // Epilogue: P1[b,k] += sum_e Y[b,e] * dX[b,e] for this wave's e-cols.
    // C layout: lane (n=l15), VGPR r -> row m = r + 8*hi16.
    int eg = ew + l15;
    float mval = mu[k * SSN + (eg & 63)];
#pragma unroll
    for (int r = 0; r < 8; ++r) {
        int b = b0 + r + 8 * hi16;
        float dx = X[(size_t)b * DD + eg] - mval;
        float p = c[r] * dx;
        p += __shfl_xor(p, 1);
        p += __shfl_xor(p, 2);
        p += __shfl_xor(p, 4);
        p += __shfl_xor(p, 8);   // half-wave sum over 16 e-cols
        if (l15 == 0) unsafeAtomicAdd(&P1[b * KC + k], p);
    }
}

// ---------------------------------------------------------------------------
// Kernel 4a: LU panel (NB cols) with partial pivoting; one block per cluster.
// Accumulates log|U_ii| into P2[k].
// ---------------------------------------------------------------------------
__global__ __launch_bounds__(256) void lu_panel_kernel(float* __restrict__ mat,
                                                       float* __restrict__ P2, int j0) {
    int k = blockIdx.x;
    float* M = mat + (size_t)k * DD * DD;
    int tid = threadIdx.x;
    __shared__ float smax[256];
    __shared__ int   simax[256];
    __shared__ int   spiv;
    __shared__ float sprow[NB];
    __shared__ float slog;
    if (tid == 0) slog = 0.0f;

    for (int cc = 0; cc < NB; ++cc) {
        int j = j0 + cc;
        // --- pivot search over column j, rows j..DD-1 ---
        float best = -1.0f; int bi = j;
        for (int i = j + tid; i < DD; i += 256) {
            float v = fabsf(M[(size_t)i * DD + j]);
            if (v > best) { best = v; bi = i; }
        }
        smax[tid] = best; simax[tid] = bi;
        __syncthreads();
        for (int s = 128; s > 0; s >>= 1) {
            if (tid < s && smax[tid + s] > smax[tid]) {
                smax[tid] = smax[tid + s]; simax[tid] = simax[tid + s];
            }
            __syncthreads();
        }
        if (tid == 0) spiv = simax[0];
        __syncthreads();
        int p = spiv;
        // --- full-row swap (keeps L columns consistent) ---
        if (p != j) {
            for (int col = tid; col < DD; col += 256) {
                float a = M[(size_t)j * DD + col];
                float b = M[(size_t)p * DD + col];
                M[(size_t)j * DD + col] = b;
                M[(size_t)p * DD + col] = a;
            }
        }
        __syncthreads();
        if (tid < NB) sprow[tid] = M[(size_t)j * DD + j0 + tid];
        __syncthreads();
        float piv = sprow[cc];
        if (tid == 0) slog += logf(fabsf(piv));
        float rpiv = 1.0f / piv;
        // --- scale column + rank-1 update restricted to panel ---
        for (int i = j + 1 + tid; i < DD; i += 256) {
            float l = M[(size_t)i * DD + j] * rpiv;
            M[(size_t)i * DD + j] = l;
            for (int c2 = cc + 1; c2 < NB; ++c2)
                M[(size_t)i * DD + j0 + c2] -= l * sprow[c2];
        }
        __syncthreads();
    }
    if (tid == 0) P2[k] += slog;  // single writer per k per launch
}

// ---------------------------------------------------------------------------
// Kernel 4b: TRSM  U12 = L11^{-1} * A12  (unit-lower L11 staged in LDS)
// ---------------------------------------------------------------------------
__global__ __launch_bounds__(256) void lu_trsm_kernel(float* __restrict__ mat, int j0) {
    int k = blockIdx.x;
    float* M = mat + (size_t)k * DD * DD;
    __shared__ float sL[NB][NB];
    int tid = threadIdx.x;
    for (int idx = tid; idx < NB * NB; idx += 256)
        sL[idx >> 5][idx & 31] = M[(size_t)(j0 + (idx >> 5)) * DD + j0 + (idx & 31)];
    __syncthreads();
    for (int jj = j0 + NB + tid; jj < DD; jj += 256) {
        float a[NB];
#pragma unroll
        for (int t = 0; t < NB; ++t) a[t] = M[(size_t)(j0 + t) * DD + jj];
#pragma unroll 4
        for (int r = 1; r < NB; ++r) {
            float acc = a[r];
            for (int t = 0; t < r; ++t) acc -= sL[r][t] * a[t];
            a[r] = acc;
        }
#pragma unroll
        for (int t = 1; t < NB; ++t) M[(size_t)(j0 + t) * DD + jj] = a[t];
    }
}

// ---------------------------------------------------------------------------
// Kernel 4c: trailing update A22 -= L21 * U12 via v_wmma_f32_16x16x4_f32.
// One 16x16 tile per wave; K = NB = 32 -> 8 WMMAs per tile. Full-GPU grid.
// ---------------------------------------------------------------------------
__global__ __launch_bounds__(256) void lu_update_kernel(float* __restrict__ mat,
                                                        int j0, int ntiles1d) {
    int ntiles = ntiles1d * ntiles1d;
    int g = blockIdx.x * 8 + (threadIdx.x >> 5);
    if (g >= KC * ntiles) return;
    int k = g / ntiles;
    int t = g % ntiles;
    int m0 = j0 + NB + (t / ntiles1d) * 16;
    int n0 = j0 + NB + (t % ntiles1d) * 16;
    float* M = mat + (size_t)k * DD * DD;
    int lane = threadIdx.x & 31;
    int hi = lane >> 4, l15 = lane & 15;

    v8f c;
#pragma unroll
    for (int r = 0; r < 8; ++r)
        c[r] = M[(size_t)(m0 + r + 8 * hi) * DD + n0 + l15];

    const float* arow = M + (size_t)(m0 + l15) * DD + j0;  // L21 row for lane
#pragma unroll
    for (int kk = 0; kk < NB / 4; ++kk) {
        int kbv = kk * 4 + 2 * hi;   // f32 A/B lane layout: K = 2*(lane>=16)+{0,1}
        v2f av, bv;
        av.x = -arow[kbv];
        av.y = -arow[kbv + 1];
        bv.x = M[(size_t)(j0 + kbv) * DD + n0 + l15];
        bv.y = M[(size_t)(j0 + kbv + 1) * DD + n0 + l15];
        c = __builtin_amdgcn_wmma_f32_16x16x4_f32(false, av, false, bv, (short)0, c, false, false);
    }
#pragma unroll
    for (int r = 0; r < 8; ++r)
        M[(size_t)(m0 + r + 8 * hi) * DD + n0 + l15] = c[r];
}

// ---------------------------------------------------------------------------
// Kernel 5: lle = -0.5*P1 + P2; softmax over K=32 -> exactly one wave32 per b
// ---------------------------------------------------------------------------
__global__ __launch_bounds__(256) void softmax_kernel(const float* __restrict__ P1,
                                                      const float* __restrict__ P2,
                                                      float* __restrict__ out) {
    int b = blockIdx.x * 8 + (threadIdx.x >> 5);
    int lane = threadIdx.x & 31;
    float lle = -0.5f * P1[b * KC + lane] + P2[lane];
    float m = lle;
    m = fmaxf(m, __shfl_xor(m, 1));
    m = fmaxf(m, __shfl_xor(m, 2));
    m = fmaxf(m, __shfl_xor(m, 4));
    m = fmaxf(m, __shfl_xor(m, 8));
    m = fmaxf(m, __shfl_xor(m, 16));
    float e = __expf(lle - m);
    float s = e;
    s += __shfl_xor(s, 1);
    s += __shfl_xor(s, 2);
    s += __shfl_xor(s, 4);
    s += __shfl_xor(s, 8);
    s += __shfl_xor(s, 16);
    out[b * KC + lane] = e / s;
}

// ---------------------------------------------------------------------------
extern "C" void kernel_launch(void* const* d_in, const int* in_sizes, int n_in,
                              void* d_out, int out_size, void* d_ws, size_t ws_size,
                              hipStream_t stream) {
    (void)in_sizes; (void)n_in; (void)out_size; (void)ws_size;
    const float* X  = (const float*)d_in[0];   // [B,T,S]   f32
    const float* mu = (const float*)d_in[1];   // [K,S]     f32
    const float* A  = (const float*)d_in[2];   // [T,K,S,S] f32
    float*  mat  = (float*)d_ws;                         // K*D*D f32 (128 MiB)
    __bf16* matH = (__bf16*)(mat + (size_t)KC * DD * DD);   // 64 MiB
    __bf16* matL = matH + (size_t)KC * DD * DD;             // 64 MiB
    float*  P1   = (float*)(matL + (size_t)KC * DD * DD);   // [B,K]
    float*  P2   = P1 + BATCH * KC;                         // [K]
    float*  out  = (float*)d_out;                           // [B,K]

    build_mat_kernel<<<KC * TT * TT, 256, 0, stream>>>(A, mat, matH, matL);
    init_kernel<<<(BATCH * KC + 255) / 256, 256, 0, stream>>>(P1, P2);

    // Quadratic form (dominant 34 GFLOP): 32 k * 32 b-tiles * 8 e-macros
    quadform_kernel<<<KC * (BATCH / 16) * (DD / 128), 256, 0, stream>>>(X, mu, matH, matL, P1);

    // Blocked LU with partial pivoting (destroys mat; runs after GEMM)
    for (int p = 0; p < DD / NB; ++p) {
        int j0 = p * NB;
        lu_panel_kernel<<<KC, 256, 0, stream>>>(mat, P2, j0);
        int M2 = DD - j0 - NB;
        if (M2 > 0) {
            lu_trsm_kernel<<<KC, 256, 0, stream>>>(mat, j0);
            int n1 = M2 / 16;
            int waves = KC * n1 * n1;
            lu_update_kernel<<<(waves + 7) / 8, 256, 0, stream>>>(mat, j0, n1);
        }
    }

    softmax_kernel<<<BATCH / 8, 256, 0, stream>>>(P1, P2, out);
}
